// CLM_23038204575917
// MI455X (gfx1250) — matrix operand order repaired and checked by reading, not compiled, not run on hardware
//
#include <hip/hip_runtime.h>

// ---------------------------------------------------------------------------
// GPT forward (B=2, T=1024, E=1024, H=16, HD=64, L=4, FFD=4096, V=32000)
// MI455X / gfx1250: wave32, bf16 WMMA (v_wmma_f32_16x16x32_bf16), f32 accum.
// ---------------------------------------------------------------------------

typedef float  v8f   __attribute__((ext_vector_type(8)));
typedef __bf16 v16bf __attribute__((ext_vector_type(16)));

#define E_   1024
#define T_   1024
#define H_   16
#define HD_  64
#define L_   4
#define FFD_ 4096
#define V_   32000
#define B_   2
#define M_   (B_ * T_)

// Native hardware converts (let clang pick v_cvt/bf16 datapath on gfx1250).
__device__ __forceinline__ unsigned short f2bf(float f) {
  union { __bf16 b; unsigned short u; } cv;
  cv.b = (__bf16)f;
  return cv.u;
}
__device__ __forceinline__ float bf2f(unsigned short s) {
  union { unsigned short u; __bf16 b; } cv;
  cv.u = s;
  return (float)cv.b;
}

union BF16Frag { unsigned short u[16]; v16bf v; };
union AccFrag  { float f[8];           v8f  v; };

__device__ __forceinline__ v8f wmma_bf16(v16bf a, v16bf b, v8f c) {
  return __builtin_amdgcn_wmma_f32_16x16x32_bf16(
      /*neg_a=*/false, a, /*neg_b=*/false, b,
      /*c_mod=*/(short)0, c, /*reuse_a=*/false, /*reuse_b=*/false);
}

// ---------------------------------------------------------------------------
// Generic GEMM: C[M,N] = epilogue( scale * A[M,K] @ W[N,K]^T )
// epilogue (compile-time): (+bias[N]) -> (relu) -> (+resid[M,N])
// Block: 256 threads = 8 waves; tile 128(M) x 64(N); K step 64.
// Waves in 4(M) x 2(N); each wave: 32 rows x 32 cols = 2x2 accumulators,
// 8 WMMAs per barrier pair (two K=32 chunks).
// A/W staged fp32->bf16 into LDS (24 KB); next K-tile prefetched
// (global_prefetch_b8 on gfx1250).
// ---------------------------------------------------------------------------
template <bool BIAS, bool RELU, bool RESID>
__global__ __launch_bounds__(256) void gemm_bf16_wmma(
    const float* __restrict__ A, const float* __restrict__ W,
    const float* __restrict__ bias, const float* __restrict__ resid,
    float* __restrict__ C, int M, int N, int K, float scale)
{
  __shared__ unsigned short sA[128][64];
  __shared__ unsigned short sB[64][64];

  const int tid   = threadIdx.x;
  const int lane  = tid & 31;
  const int wave  = tid >> 5;
  const int waveM = wave >> 1;     // 0..3 -> 32 rows each
  const int waveN = wave & 1;      // 0..1 -> 32 cols each
  const int colL  = lane & 15;
  const int hi    = lane >> 4;     // 0 or 1

  const int m0 = blockIdx.y * 128;
  const int n0 = blockIdx.x * 64;

  v8f acc[2][2];
#pragma unroll
  for (int mi = 0; mi < 2; ++mi)
#pragma unroll
    for (int ni = 0; ni < 2; ++ni)
      acc[mi][ni] = (v8f){0.f,0.f,0.f,0.f,0.f,0.f,0.f,0.f};

  // A loader: 128x64 = 8192 elems, 32 contiguous floats per thread.
  const int aRow = tid >> 1;          // 0..127
  const int aCol = (tid & 1) * 32;    // 0 or 32
  // B loader: 64x64 = 4096 elems, 16 contiguous floats per thread.
  const int bRow = tid >> 2;          // 0..63
  const int bCol = (tid & 3) * 16;    // 0,16,32,48

  const float* ap = A + (size_t)(m0 + aRow) * K + aCol;
  const float* wp = W + (size_t)(n0 + bRow) * K + bCol;

  for (int k0 = 0; k0 < K; k0 += 64) {
#pragma unroll
    for (int f4 = 0; f4 < 8; ++f4) {
      const float4 av = *(const float4*)(ap + f4 * 4);
      sA[aRow][aCol + f4 * 4 + 0] = f2bf(av.x);
      sA[aRow][aCol + f4 * 4 + 1] = f2bf(av.y);
      sA[aRow][aCol + f4 * 4 + 2] = f2bf(av.z);
      sA[aRow][aCol + f4 * 4 + 3] = f2bf(av.w);
    }
#pragma unroll
    for (int f4 = 0; f4 < 4; ++f4) {
      const float4 wv = *(const float4*)(wp + f4 * 4);
      sB[bRow][bCol + f4 * 4 + 0] = f2bf(wv.x);
      sB[bRow][bCol + f4 * 4 + 1] = f2bf(wv.y);
      sB[bRow][bCol + f4 * 4 + 2] = f2bf(wv.z);
      sB[bRow][bCol + f4 * 4 + 3] = f2bf(wv.w);
    }
    ap += 64; wp += 64;
    if (k0 + 64 < K) {                 // prefetch next K tile into cache
      __builtin_prefetch(ap, 0, 1);
      __builtin_prefetch(wp, 0, 1);
    }
    __syncthreads();

#pragma unroll
    for (int c = 0; c < 2; ++c) {      // two K=32 chunks
      // A fragments: lane<16 -> K 0..15; lane>=16 -> K 16..31
      BF16Frag a[2];
#pragma unroll
      for (int mi = 0; mi < 2; ++mi) {
        const int ar = waveM * 32 + mi * 16 + colL;
#pragma unroll
        for (int i = 0; i < 16; ++i) {
          const int kk = c * 32 + (hi ? 16 : 0) + i;
          a[mi].u[i] = sA[ar][kk];
        }
      }
      // B fragments: lane<16 -> K 0..15; lane>=16 -> K 16..31
      BF16Frag b[2];
      const int kb = c * 32 + (hi ? 16 : 0);
#pragma unroll
      for (int ni = 0; ni < 2; ++ni) {
        const int br = waveN * 32 + ni * 16 + colL;
#pragma unroll
        for (int i = 0; i < 16; ++i) b[ni].u[i] = sB[br][kb + i];
      }
#pragma unroll
      for (int mi = 0; mi < 2; ++mi)
#pragma unroll
        for (int ni = 0; ni < 2; ++ni)
          acc[mi][ni] = wmma_bf16(a[mi].v, b[ni].v, acc[mi][ni]);
    }
    __syncthreads();
  }

  // Straight-line epilogue (flags compile-time).
#pragma unroll
  for (int mi = 0; mi < 2; ++mi)
#pragma unroll
    for (int ni = 0; ni < 2; ++ni) {
      AccFrag af; af.v = acc[mi][ni];
      const int col  = n0 + waveN * 32 + ni * 16 + colL;
      const int row0 = m0 + waveM * 32 + mi * 16 + hi * 8;
      float bcol = 0.f;
      if (BIAS) bcol = bias[col];
#pragma unroll
      for (int r = 0; r < 8; ++r) {
        const size_t ci = (size_t)(row0 + r) * N + col;
        float vv = af.f[r] * scale;
        if (BIAS)  vv += bcol;
        if (RELU)  vv = fmaxf(vv, 0.f);
        if (RESID) vv += resid[ci];
        C[ci] = vv;
      }
    }
}

// ---------------------------------------------------------------------------
// Causal attention, one wave per (b, head, 16-query block).
// q,k,v: [B,T,E] fp32, head h occupies columns [h*64, h*64+64).
// Phase 1: S = QK^T * HD^-0.5 via WMMA (two K=32 chunks), causal mask,
//          bf16 scores into LDS (16 x 1024 = 32 KB).
// Phase 2: row softmax (lanes 0..15, one row each).
// Phase 3: O = P V via WMMA over 32-key chunks, 4 N-tiles of 16 dims.
// ---------------------------------------------------------------------------
__global__ __launch_bounds__(32) void attn_wmma(
    const float* __restrict__ q, const float* __restrict__ k,
    const float* __restrict__ v, float* __restrict__ o)
{
  __shared__ unsigned short sc[16][1024];

  const int lane = threadIdx.x;
  const int colL = lane & 15;
  const int hi   = lane >> 4;
  const int qb   = blockIdx.x;        // 0..63
  const int hd   = blockIdx.y;        // head
  const int bb   = blockIdx.z;        // batch
  const int t0   = qb * 16;

  // Q fragment is invariant across key blocks: build once for both K-chunks.
  BF16Frag aq[2];
  const float* qp = q + ((size_t)bb * T_ + t0 + colL) * E_ + hd * HD_;
#pragma unroll
  for (int ch = 0; ch < 2; ++ch)
#pragma unroll
    for (int i = 0; i < 16; ++i) {
      const int dd = ch * 32 + (hi ? 16 : 0) + i;
      aq[ch].u[i] = f2bf(qp[dd]);
    }

  for (int kb = 0; kb <= qb; ++kb) {
    v8f acc = {0.f,0.f,0.f,0.f,0.f,0.f,0.f,0.f};
    const float* kp = k + ((size_t)bb * T_ + kb * 16 + colL) * E_ + hd * HD_;
#pragma unroll
    for (int ch = 0; ch < 2; ++ch) {
      BF16Frag bf;
#pragma unroll
      for (int i = 0; i < 16; ++i)
        bf.u[i] = f2bf(kp[ch * 32 + (hi ? 16 : 0) + i]);
      acc = wmma_bf16(aq[ch].v, bf.v, acc);
    }
    AccFrag af; af.v = acc;
#pragma unroll
    for (int r = 0; r < 8; ++r) {
      const int row = r + hi * 8;        // query row within block
      const int j   = kb * 16 + colL;    // key index
      float sv = af.f[r] * 0.125f;       // HD^-0.5
      if (j > t0 + row) sv = -1e30f;     // causal mask
      sc[row][j] = f2bf(sv);
    }
  }
  __syncthreads();

  const int nkb  = qb + 1;
  const int nkc  = (nkb + 1) >> 1;       // 32-key chunks
  const int npad = nkc * 32;

  if (lane < 16) {
    const int row   = lane;
    const int limit = t0 + row;          // inclusive last valid key
    float mx = -1e30f;
    for (int j = 0; j <= limit; ++j) mx = fmaxf(mx, bf2f(sc[row][j]));
    float sum = 0.f;
    for (int j = 0; j <= limit; ++j) {
      const float e = expf(bf2f(sc[row][j]) - mx);
      sum += e;
      sc[row][j] = f2bf(e);
    }
    const float inv = 1.f / sum;
    for (int j = 0; j <= limit; ++j) sc[row][j] = f2bf(bf2f(sc[row][j]) * inv);
    for (int j = limit + 1; j < npad; ++j) sc[row][j] = 0;   // pad P with zeros
  }
  __syncthreads();

  v8f oacc[4];
#pragma unroll
  for (int nt = 0; nt < 4; ++nt) oacc[nt] = (v8f){0.f,0.f,0.f,0.f,0.f,0.f,0.f,0.f};

  for (int kc = 0; kc < nkc; ++kc) {
    BF16Frag a;                          // P rows, keys as K-dim
#pragma unroll
    for (int i = 0; i < 16; ++i) {
      const int jj = kc * 32 + (hi ? 16 : 0) + i;
      a.u[i] = sc[colL][jj];
    }
#pragma unroll
    for (int nt = 0; nt < 4; ++nt) {
      BF16Frag bf;                       // V: 32 keys x 16 dims
#pragma unroll
      for (int i = 0; i < 16; ++i) {
        const int j = kc * 32 + (hi ? 16 : 0) + i;
        bf.u[i] = f2bf(v[((size_t)bb * T_ + j) * E_ + hd * HD_ + nt * 16 + colL]);
      }
      oacc[nt] = wmma_bf16(a.v, bf.v, oacc[nt]);
    }
  }

#pragma unroll
  for (int nt = 0; nt < 4; ++nt) {
    AccFrag af; af.v = oacc[nt];
#pragma unroll
    for (int r = 0; r < 8; ++r) {
      const int row = t0 + r + hi * 8;
      o[((size_t)bb * T_ + row) * E_ + hd * HD_ + nt * 16 + colL] = af.f[r];
    }
  }
}

// ---------------------------------------------------------------------------
// LayerNorm: one block (256 thr) per row of [M, E].
// ---------------------------------------------------------------------------
__global__ __launch_bounds__(256) void ln_kernel(
    const float* __restrict__ x, const float* __restrict__ g,
    const float* __restrict__ bta, float* __restrict__ y)
{
  __shared__ float s1[256], s2[256];
  const int row = blockIdx.x, tid = threadIdx.x;
  const float* xr = x + (size_t)row * E_;
  float s = 0.f, ss = 0.f;
  for (int e = tid; e < E_; e += 256) { const float vv = xr[e]; s += vv; ss += vv * vv; }
  s1[tid] = s; s2[tid] = ss; __syncthreads();
  for (int st = 128; st > 0; st >>= 1) {
    if (tid < st) { s1[tid] += s1[tid + st]; s2[tid] += s2[tid + st]; }
    __syncthreads();
  }
  const float mu  = s1[0] * (1.f / E_);
  const float var = s2[0] * (1.f / E_) - mu * mu;
  const float inv = rsqrtf(var + 1e-5f);
  float* yr = y + (size_t)row * E_;
  for (int e = tid; e < E_; e += 256)
    yr[e] = (xr[e] - mu) * inv * g[e] + bta[e];
}

// ---------------------------------------------------------------------------
// Embedding: x[b,t,:] = tok_emb[idx[b,t],:] + pos_emb[t,:]
// ---------------------------------------------------------------------------
__global__ __launch_bounds__(256) void embed_kernel(
    const int* __restrict__ idx, const float* __restrict__ tok,
    const float* __restrict__ pos, float* __restrict__ x)
{
  const int row = blockIdx.x;            // 0..M-1
  const int t   = row % T_;
  const int tk  = idx[row];
  const float* tp = tok + (size_t)tk * E_;
  const float* pp = pos + (size_t)t  * E_;
  float* xr = x + (size_t)row * E_;
  for (int e = threadIdx.x; e < E_; e += 256) xr[e] = tp[e] + pp[e];
}

// ---------------------------------------------------------------------------
// Host orchestration
// ---------------------------------------------------------------------------
extern "C" void kernel_launch(void* const* d_in, const int* in_sizes, int n_in,
                              void* d_out, int out_size, void* d_ws, size_t ws_size,
                              hipStream_t stream)
{
  (void)in_sizes; (void)n_in; (void)out_size; (void)ws_size;
  const int*   idx     = (const int*)  d_in[0];
  const float* tok_emb = (const float*)d_in[1];
  const float* pos_emb = (const float*)d_in[2];
  const float* Wq      = (const float*)d_in[3];
  const float* Wk      = (const float*)d_in[4];
  const float* Wv      = (const float*)d_in[5];
  const float* Wp      = (const float*)d_in[6];
  const float* ln1_g   = (const float*)d_in[7];
  const float* ln1_b   = (const float*)d_in[8];
  const float* W1      = (const float*)d_in[9];
  const float* b1      = (const float*)d_in[10];
  const float* W2      = (const float*)d_in[11];
  const float* b2      = (const float*)d_in[12];
  const float* ln2_g   = (const float*)d_in[13];
  const float* ln2_b   = (const float*)d_in[14];
  const float* lnf_g   = (const float*)d_in[15];
  const float* lnf_b   = (const float*)d_in[16];
  const float* head_b  = (const float*)d_in[17];

  float* ws = (float*)d_ws;
  float* x  = ws;                          // [M,E] residual stream
  float* h  = x + (size_t)M_ * E_;         // [M,E] ln out / attn out (reused)
  float* q  = h + (size_t)M_ * E_;         // [M,E]
  float* k  = q + (size_t)M_ * E_;         // [M,E]
  float* v  = k + (size_t)M_ * E_;         // [M,E]
  float* ff = v + (size_t)M_ * E_;         // [M,FFD]

  const float sE = 0.03125f;               // E^-0.5 (qkv scale and proj scale)
  const dim3 gE(E_ / 64, M_ / 128);        // N-tiles x M-tiles

  embed_kernel<<<M_, 256, 0, stream>>>(idx, tok_emb, pos_emb, x);

  for (int l = 0; l < L_; ++l) {
    ln_kernel<<<M_, 256, 0, stream>>>(x, ln1_g + (size_t)l * E_, ln1_b + (size_t)l * E_, h);
    gemm_bf16_wmma<false,false,false><<<gE, 256, 0, stream>>>(
        h, Wq + (size_t)l * E_ * E_, nullptr, nullptr, q, M_, E_, E_, sE);
    gemm_bf16_wmma<false,false,false><<<gE, 256, 0, stream>>>(
        h, Wk + (size_t)l * E_ * E_, nullptr, nullptr, k, M_, E_, E_, sE);
    gemm_bf16_wmma<false,false,false><<<gE, 256, 0, stream>>>(
        h, Wv + (size_t)l * E_ * E_, nullptr, nullptr, v, M_, E_, E_, sE);
    attn_wmma<<<dim3(T_ / 16, H_, B_), 32, 0, stream>>>(q, k, v, h);
    gemm_bf16_wmma<false,false,true><<<gE, 256, 0, stream>>>(
        h, Wp + (size_t)l * E_ * E_, nullptr, x, x, M_, E_, E_, sE);
    ln_kernel<<<M_, 256, 0, stream>>>(x, ln2_g + (size_t)l * E_, ln2_b + (size_t)l * E_, h);
    gemm_bf16_wmma<true,true,false><<<dim3(FFD_ / 64, M_ / 128), 256, 0, stream>>>(
        h, W1 + (size_t)l * FFD_ * E_, b1 + (size_t)l * FFD_, nullptr, ff, M_, FFD_, E_, 1.f);
    gemm_bf16_wmma<true,false,true><<<gE, 256, 0, stream>>>(
        ff, W2 + (size_t)l * E_ * FFD_, b2 + (size_t)l * E_, x, x, M_, E_, FFD_, 1.f);
  }

  ln_kernel<<<M_, 256, 0, stream>>>(x, lnf_g, lnf_b, h);
  gemm_bf16_wmma<true,false,false><<<dim3(V_ / 64, M_ / 128), 256, 0, stream>>>(
      h, tok_emb, head_b, nullptr, (float*)d_out, M_, V_, E_, 1.f);
}
